// KNNGraphInferenceModel_44753559224391
// MI455X (gfx1250) — matrix operand order repaired and checked by reading, not compiled, not run on hardware
//
#include <hip/hip_runtime.h>
#include <hip/hip_bf16.h>

// ---------------------------------------------------------------------------
// KNN-GNN inference for MI455X (gfx1250, wave32).
//  - KNN gram matrix via v_wmma_f32_16x16x32_bf16 with hi/lo bf16 split
//    (near-fp32 distance precision at bf16 WMMA rates), fused top-17 select.
//  - Column stream staged into double-buffered LDS via the gfx1250 async
//    Global->LDS engine (ASYNCcnt) so DMA of chunk n+1 overlaps WMMA of n.
//  - Edge/node MLPs with training-mode BatchNorm via deterministic two-stage
//    per-feature stat reductions; scatter-add via global f32 atomics.
// ---------------------------------------------------------------------------

#define NN   16384
#define DD   64
#define KNN  17
#define EE   (NN * 16)          // 262144 edges
#define EPSB 1e-5f

typedef __attribute__((ext_vector_type(16))) __bf16 v16bf;
typedef __attribute__((ext_vector_type(8)))  float  v8f;
typedef int v4i_t __attribute__((vector_size(16)));   // matches builtin param

union AU { unsigned short u[16]; v16bf v; };

#define GAS __attribute__((address_space(1)))
#define LAS __attribute__((address_space(3)))

#if defined(__HIP_DEVICE_COMPILE__) && __has_builtin(__builtin_amdgcn_global_load_async_to_lds_b128)
#define USE_ASYNC 1
#else
#define USE_ASYNC 0
#endif

static __device__ __forceinline__ unsigned short f2bf(float x) {
    unsigned int u = __float_as_uint(x);
    u += 0x7FFFu + ((u >> 16) & 1u);          // round-to-nearest-even
    return (unsigned short)(u >> 16);
}
static __device__ __forceinline__ float bf2f(unsigned short b) {
    return __uint_as_float(((unsigned int)b) << 16);
}

// copy one 8KB chunk (64 cols x 64 bf16) global -> LDS; 64 threads
static __device__ __forceinline__ void chunk_copy(const unsigned short* __restrict__ gsrc,
                                                  unsigned short* ldst, int tid) {
#if USE_ASYNC
    GAS v4i_t* g = (GAS v4i_t*)((GAS char*)gsrc) + tid;
    LAS v4i_t* l = (LAS v4i_t*)((LAS char*)ldst) + tid;
    #pragma unroll
    for (int it = 0; it < 8; ++it)
        __builtin_amdgcn_global_load_async_to_lds_b128(g + it * 64, l + it * 64, 0, 0);
#else
    const uint4* g = (const uint4*)gsrc;
    uint4* l = (uint4*)ldst;
    #pragma unroll
    for (int it = 0; it < 8; ++it) l[tid + it * 64] = g[tid + it * 64];
#endif
}

// ---------------------------------------------------------------------------
// h = pos @ Win + b        (N x 11) @ (11 x 64)
// ---------------------------------------------------------------------------
__global__ __launch_bounds__(256) void k_lin_in(const float* __restrict__ pos,
                                                const float* __restrict__ W,
                                                const float* __restrict__ b,
                                                float* __restrict__ h) {
    int idx = blockIdx.x * 256 + threadIdx.x;      // < N*64
    int n = idx >> 6, f = idx & 63;
    float y = b[f];
    #pragma unroll
    for (int k = 0; k < 11; ++k) y += pos[n * 11 + k] * W[k * 64 + f];
    h[idx] = y;
}

// ---------------------------------------------------------------------------
// prep: sq[n] = ||h_n||^2 ; hi/lo bf16 split of h
// ---------------------------------------------------------------------------
__global__ __launch_bounds__(256) void k_prep(const float* __restrict__ h,
                                              unsigned short* __restrict__ hhi,
                                              unsigned short* __restrict__ hlo,
                                              float* __restrict__ sq) {
    int node = blockIdx.x * 8 + (threadIdx.x >> 5);
    int lane = threadIdx.x & 31;
    float v0 = h[node * 64 + lane];
    float v1 = h[node * 64 + lane + 32];
    unsigned short h0 = f2bf(v0), h1 = f2bf(v1);
    hhi[node * 64 + lane]      = h0;
    hhi[node * 64 + lane + 32] = h1;
    hlo[node * 64 + lane]      = f2bf(v0 - bf2f(h0));
    hlo[node * 64 + lane + 32] = f2bf(v1 - bf2f(h1));
    float ss = v0 * v0 + v1 * v1;
    #pragma unroll
    for (int o = 16; o > 0; o >>= 1) ss += __shfl_xor(ss, o, 32);
    if (lane == 0) sq[node] = ss;
}

// ---------------------------------------------------------------------------
// KNN: fused WMMA gram + top-17 selection.
// block = 64 threads (2 waves). Each wave owns 32 query rows (two 16x16 C
// tiles). 64-column chunks double-buffered in LDS via async Global->LDS.
// ---------------------------------------------------------------------------
__global__ __launch_bounds__(64) void k_knn(const unsigned short* __restrict__ hhi,
                                            const unsigned short* __restrict__ hlo,
                                            const float* __restrict__ sq,
                                            int* __restrict__ knnidx) {
    __shared__ unsigned short colHi[2][64 * 64];   // double-buffered col chunk
    __shared__ unsigned short colLo[2][64 * 64];
    __shared__ float selb[2][32][64];              // per-wave d2 staging
    __shared__ float topD[2][32][KNN];
    __shared__ int   topI[2][32][KNN];

    const int tid  = threadIdx.x;
    const int w    = tid >> 5;
    const int lane = tid & 31;
    const int g    = lane >> 4;      // lane group (0/1)
    const int nn   = lane & 15;      // B/C column within tile
    const int m    = lane & 15;      // A row within tile
    const int rowBase = blockIdx.x * 64;

    #pragma unroll
    for (int j = 0; j < KNN; ++j) { topD[w][lane][j] = 3.0e38f; topI[w][lane][j] = 0; }

    // row ||.||^2 terms are loop-invariant per lane -> registers
    float srow[2][8];
    #pragma unroll
    for (int T = 0; T < 2; ++T)
        #pragma unroll
        for (int v = 0; v < 8; ++v)
            srow[T][v] = sq[rowBase + w * 32 + T * 16 + g * 8 + v];

    // ---- load A tiles (per-wave 32 rows, hi/lo, K=64 split in two k-steps)
    // 16-bit A 16x32 layout: lane grp g, element t -> K = k0+g*16+t
    // (contiguous, matching B)
    AU Ahi[2][2], Alo[2][2];
    #pragma unroll
    for (int T = 0; T < 2; ++T) {
        const int row = rowBase + w * 32 + T * 16 + m;
        const unsigned short* ph = hhi + (size_t)row * DD;
        const unsigned short* pl = hlo + (size_t)row * DD;
        #pragma unroll
        for (int ks = 0; ks < 2; ++ks) {
            #pragma unroll
            for (int t = 0; t < 16; ++t) {
                int K = ks * 32 + g * 16 + t;
                Ahi[T][ks].u[t] = ph[K];
                Alo[T][ks].u[t] = pl[K];
            }
        }
    }

    float worst = 3.0e38f;
    const int nChunks = NN / 64;

    // prologue: start DMA of chunk 0 into buffer 0
    chunk_copy(hhi, &colHi[0][0], tid);
    chunk_copy(hlo, &colLo[0][0], tid);

    for (int ch = 0; ch < nChunks; ++ch) {
        const int buf = ch & 1;
        const int c0  = ch * 64;
#if USE_ASYNC
        asm volatile("s_wait_asynccnt 0x0" ::: "memory");  // my chunk-ch DMA done
#endif
        __syncthreads();                                   // all waves' DMA done
        if (ch + 1 < nChunks) {                            // overlap: DMA ch+1
            chunk_copy(hhi + (size_t)(c0 + 64) * DD, &colHi[buf ^ 1][0], tid);
            chunk_copy(hlo + (size_t)(c0 + 64) * DD, &colLo[buf ^ 1][0], tid);
#if !USE_ASYNC
            __builtin_prefetch(hhi + (size_t)(c0 + 128) * DD + tid * 64, 0, 0);
#endif
        }

        // ---- compute 4 column tiles of 16
        #pragma unroll 1
        for (int ct = 0; ct < 4; ++ct) {
            // B 32x16 bf16 layout: lanes 0-15 K=k0..k0+15, lanes 16-31 K=+16;
            // per lane 16 contiguous bf16 of column (ct*16+nn)
            v16bf Bh[2], Bl[2];
            #pragma unroll
            for (int ks = 0; ks < 2; ++ks) {
                const int col = ct * 16 + nn;
                Bh[ks] = *(const v16bf*)&colHi[buf][col * 64 + ks * 32 + g * 16];
                Bl[ks] = *(const v16bf*)&colLo[buf][col * 64 + ks * 32 + g * 16];
            }
            const float sc = sq[c0 + ct * 16 + nn];        // L2-hot scalar
            #pragma unroll
            for (int T = 0; T < 2; ++T) {
                v8f acc = {};
                #pragma unroll
                for (int ks = 0; ks < 2; ++ks) {
                    acc = __builtin_amdgcn_wmma_f32_16x16x32_bf16(
                        false, Ahi[T][ks].v, false, Bh[ks], (short)0, acc, false, false);
                    acc = __builtin_amdgcn_wmma_f32_16x16x32_bf16(
                        false, Ahi[T][ks].v, false, Bl[ks], (short)0, acc, false, false);
                    acc = __builtin_amdgcn_wmma_f32_16x16x32_bf16(
                        false, Alo[T][ks].v, false, Bh[ks], (short)0, acc, false, false);
                }
                // C layout: VGPR v, lane grp g -> row M = g*8+v, col = nn
                #pragma unroll
                for (int v = 0; v < 8; ++v) {
                    const int rl = T * 16 + g * 8 + v;
                    float d2 = srow[T][v] + sc - 2.0f * acc[v];
                    selb[w][rl][ct * 16 + nn] = d2;
                }
            }
        }
        __syncthreads();

        // ---- selection: lane owns row `lane`; threshold-guarded insert
        #pragma unroll 1
        for (int c = 0; c < 64; ++c) {
            float d = fmaxf(selb[w][lane][c], 0.0f);
            if (d < worst) {
                int gcol = c0 + c;
                int p = KNN - 1;
                while (p > 0 && d < topD[w][lane][p - 1]) {
                    topD[w][lane][p] = topD[w][lane][p - 1];
                    topI[w][lane][p] = topI[w][lane][p - 1];
                    --p;
                }
                topD[w][lane][p] = d;
                topI[w][lane][p] = gcol;
                worst = topD[w][lane][KNN - 1];
            }
        }
    }

    const int rowG = rowBase + w * 32 + lane;
    #pragma unroll
    for (int j = 0; j < KNN; ++j) knnidx[rowG * KNN + j] = topI[w][lane][j];
}

// ---------------------------------------------------------------------------
// P = h @ W1[0:64,:] ; Q = h @ W1[64:128,:]   (factor W1 out of per-edge work)
// ---------------------------------------------------------------------------
__global__ __launch_bounds__(64) void k_pq(const float* __restrict__ h,
                                           const float* __restrict__ W1,
                                           float* __restrict__ P,
                                           float* __restrict__ Q) {
    int f = threadIdx.x;
    int n0 = blockIdx.x * 16;
    for (int r = 0; r < 16; ++r) {
        int n = n0 + r;
        float p = 0.f, q = 0.f;
        #pragma unroll 8
        for (int k = 0; k < 64; ++k) {
            float hv = h[n * 64 + k];
            p += hv * W1[k * 64 + f];
            q += hv * W1[(64 + k) * 64 + f];
        }
        P[n * 64 + f] = p;
        Q[n * 64 + f] = q;
    }
}

// ---------------------------------------------------------------------------
// Edge stats for BN1: y1[e] = P[to]+Q[frm]+b1 ; per-block partial sums
// ---------------------------------------------------------------------------
__global__ __launch_bounds__(64) void k_mstats1(const float* __restrict__ P,
                                                const float* __restrict__ Q,
                                                const int* __restrict__ knnidx,
                                                const float* __restrict__ b1,
                                                float* __restrict__ part) {
    int f = threadIdx.x;
    float bb = b1[f];
    float s1 = 0.f, s2 = 0.f;
    int base = blockIdx.x * (EE / 512);
    for (int e = base; e < base + EE / 512; ++e) {
        int i = e & (NN - 1);
        int q = e >> 14;
        int to  = knnidx[i * KNN + q + 1];
        int frm = knnidx[i * KNN];
        float y = P[to * 64 + f] + Q[frm * 64 + f] + bb;
        s1 += y; s2 += y * y;
    }
    part[blockIdx.x * 128 + f]      = s1;
    part[blockIdx.x * 128 + 64 + f] = s2;
}

// ---------------------------------------------------------------------------
// Reduce per-feature partials -> BN affine (a, c): z = relu(a*y + c)
// ---------------------------------------------------------------------------
__global__ __launch_bounds__(64) void k_reduce_bn(const float* __restrict__ part,
                                                  int nb, float cnt,
                                                  const float* __restrict__ g,
                                                  const float* __restrict__ be,
                                                  float* __restrict__ bnp) {
    int f = threadIdx.x;
    float s1 = 0.f, s2 = 0.f;
    for (int b = 0; b < nb; ++b) {
        s1 += part[b * 128 + f];
        s2 += part[b * 128 + 64 + f];
    }
    float mean = s1 / cnt;
    float var  = s2 / cnt - mean * mean;
    float a = g[f] * rsqrtf(var + EPSB);
    bnp[f]      = a;
    bnp[64 + f] = be[f] - a * mean;
}

// ---------------------------------------------------------------------------
// Edge MLP second half. mode 0: accumulate y2 stats. mode 1: scatter relu(bn2)
// ---------------------------------------------------------------------------
__global__ __launch_bounds__(256) void k_msg_pass(const float* __restrict__ P,
                                                  const float* __restrict__ Q,
                                                  const int* __restrict__ knnidx,
                                                  const float* __restrict__ b1,
                                                  const float* __restrict__ bnpA,
                                                  const float* __restrict__ W2,
                                                  const float* __restrict__ b2,
                                                  const float* __restrict__ bnpB,
                                                  float* __restrict__ part,
                                                  float* __restrict__ agg,
                                                  int mode) {
    __shared__ float W2s[64 * 64];
    __shared__ float zs[4][64];
    __shared__ float red[4][64];
    int tid = threadIdx.x;
    int el = tid >> 6, f = tid & 63;
    #pragma unroll
    for (int it = 0; it < 16; ++it) W2s[tid + it * 256] = W2[tid + it * 256];
    float a1 = bnpA[f], c1 = bnpA[64 + f], bb1 = b1[f], bb2 = b2[f];
    float a2 = 0.f, c2 = 0.f;
    if (mode == 1) { a2 = bnpB[f]; c2 = bnpB[64 + f]; }
    float s1 = 0.f, s2 = 0.f;
    __syncthreads();

    int base = blockIdx.x * 256;
    for (int it = 0; it < 64; ++it) {
        int e = base + it * 4 + el;
        int i = e & (NN - 1);
        int q = e >> 14;
        int to  = knnidx[i * KNN + q + 1];
        int frm = knnidx[i * KNN];
        float y1 = P[to * 64 + f] + Q[frm * 64 + f] + bb1;
        zs[el][f] = fmaxf(a1 * y1 + c1, 0.f);
        __syncthreads();
        float y2 = bb2;
        #pragma unroll 16
        for (int k = 0; k < 64; ++k) y2 += zs[el][k] * W2s[k * 64 + f];
        if (mode == 0) { s1 += y2; s2 += y2 * y2; }
        else {
            float z2 = fmaxf(a2 * y2 + c2, 0.f);
            atomicAdd(&agg[(size_t)to * 64 + f], z2);
        }
        __syncthreads();
    }
    if (mode == 0) {
        red[el][f] = s1; __syncthreads();
        if (el == 0) part[blockIdx.x * 128 + f] = red[0][f] + red[1][f] + red[2][f] + red[3][f];
        __syncthreads();
        red[el][f] = s2; __syncthreads();
        if (el == 0) part[blockIdx.x * 128 + 64 + f] = red[0][f] + red[1][f] + red[2][f] + red[3][f];
    }
}

__global__ __launch_bounds__(256) void k_zero(float* __restrict__ p, int n) {
    int i = blockIdx.x * 256 + threadIdx.x;
    if (i < n) p[i] = 0.f;
}

// ---------------------------------------------------------------------------
// Update MLP first GEMM: y1u = [h, agg] @ W1u + b1 ; stats.
// ---------------------------------------------------------------------------
__global__ __launch_bounds__(64) void k_ustats1(const float* __restrict__ h,
                                                const float* __restrict__ agg,
                                                const float* __restrict__ W1,
                                                const float* __restrict__ b1,
                                                float* __restrict__ y1u,
                                                float* __restrict__ part) {
    int f = threadIdx.x;
    float bb = b1[f];
    float s1 = 0.f, s2 = 0.f;
    int n0 = blockIdx.x * (NN / 512);
    for (int n = n0; n < n0 + NN / 512; ++n) {
        float y = bb;
        #pragma unroll 8
        for (int k = 0; k < 64; ++k) {
            y += h[n * 64 + k]   * W1[k * 64 + f];
            y += agg[n * 64 + k] * W1[(64 + k) * 64 + f];
        }
        y1u[n * 64 + f] = y;
        s1 += y; s2 += y * y;
    }
    part[blockIdx.x * 128 + f]      = s1;
    part[blockIdx.x * 128 + 64 + f] = s2;
}

// ---------------------------------------------------------------------------
// Update MLP second GEMM: y2u = relu(bn1(y1u)) @ W2u + b2 ; stats.
// ---------------------------------------------------------------------------
__global__ __launch_bounds__(256) void k_ustats2(const float* __restrict__ y1u,
                                                 const float* __restrict__ bnpA,
                                                 const float* __restrict__ W2,
                                                 const float* __restrict__ b2,
                                                 float* __restrict__ y2u,
                                                 float* __restrict__ part) {
    __shared__ float W2s[64 * 64];
    __shared__ float zs[4][64];
    __shared__ float red[4][64];
    int tid = threadIdx.x;
    int el = tid >> 6, f = tid & 63;
    #pragma unroll
    for (int it = 0; it < 16; ++it) W2s[tid + it * 256] = W2[tid + it * 256];
    float a1 = bnpA[f], c1 = bnpA[64 + f], bb2 = b2[f];
    float s1 = 0.f, s2 = 0.f;
    __syncthreads();

    int base = blockIdx.x * 64;
    for (int it = 0; it < 16; ++it) {
        int n = base + it * 4 + el;
        float y1 = y1u[n * 64 + f];
        zs[el][f] = fmaxf(a1 * y1 + c1, 0.f);
        __syncthreads();
        float y2 = bb2;
        #pragma unroll 16
        for (int k = 0; k < 64; ++k) y2 += zs[el][k] * W2s[k * 64 + f];
        y2u[n * 64 + f] = y2;
        s1 += y2; s2 += y2 * y2;
        __syncthreads();
    }
    red[el][f] = s1; __syncthreads();
    if (el == 0) part[blockIdx.x * 128 + f] = red[0][f] + red[1][f] + red[2][f] + red[3][f];
    __syncthreads();
    red[el][f] = s2; __syncthreads();
    if (el == 0) part[blockIdx.x * 128 + 64 + f] = red[0][f] + red[1][f] + red[2][f] + red[3][f];
}

__global__ __launch_bounds__(256) void k_ufinal(float* __restrict__ h,
                                                const float* __restrict__ y2u,
                                                const float* __restrict__ bnpB) {
    int idx = blockIdx.x * 256 + threadIdx.x;   // < N*64
    int f = idx & 63;
    h[idx] += fmaxf(bnpB[f] * y2u[idx] + bnpB[64 + f], 0.f);
}

// ---------------------------------------------------------------------------
// global mean pool + prediction head
// ---------------------------------------------------------------------------
__global__ __launch_bounds__(64) void k_pool1(const float* __restrict__ h,
                                              float* __restrict__ part) {
    int f = threadIdx.x;
    float s = 0.f;
    int n0 = blockIdx.x * 64;
    for (int r = 0; r < 64; ++r) s += h[(n0 + r) * 64 + f];
    part[blockIdx.x * 64 + f] = s;
}

__global__ __launch_bounds__(64) void k_pool2(const float* __restrict__ part,
                                              const float* __restrict__ predW,
                                              const float* __restrict__ predB,
                                              float* __restrict__ out) {
    __shared__ float r[64];
    int f = threadIdx.x;
    float s = 0.f;
    for (int b = 0; b < 256; ++b) s += part[b * 64 + f];
    r[f] = (s / (float)NN) * predW[f];
    __syncthreads();
    #pragma unroll
    for (int o = 32; o > 0; o >>= 1) {
        if (f < o) r[f] += r[f + o];
        __syncthreads();
    }
    if (f == 0) out[0] = r[0] + predB[0];
}

// ---------------------------------------------------------------------------
extern "C" void kernel_launch(void* const* d_in, const int* in_sizes, int n_in,
                              void* d_out, int out_size, void* d_ws, size_t ws_size,
                              hipStream_t stream) {
    (void)in_sizes; (void)n_in; (void)out_size; (void)ws_size;
    const float* pos   = (const float*)d_in[0];
    const float* linW  = (const float*)d_in[1];
    const float* linB  = (const float*)d_in[2];
    const float* predW = (const float*)d_in[3];
    const float* predB = (const float*)d_in[4];
    const float* msgW1 = (const float*)d_in[5];
    const float* msgB1 = (const float*)d_in[6];
    const float* msgG1 = (const float*)d_in[7];
    const float* msgBe1= (const float*)d_in[8];
    const float* msgW2 = (const float*)d_in[9];
    const float* msgB2 = (const float*)d_in[10];
    const float* msgG2 = (const float*)d_in[11];
    const float* msgBe2= (const float*)d_in[12];
    const float* updW1 = (const float*)d_in[13];
    const float* updB1 = (const float*)d_in[14];
    const float* updG1 = (const float*)d_in[15];
    const float* updBe1= (const float*)d_in[16];
    const float* updW2 = (const float*)d_in[17];
    const float* updB2 = (const float*)d_in[18];
    const float* updG2 = (const float*)d_in[19];
    const float* updBe2= (const float*)d_in[20];

    char* w = (char*)d_ws;
    size_t o = 0;
    const size_t F32MAT = (size_t)NN * 64 * sizeof(float);       // 4 MB
    float* h    = (float*)(w + o); o += F32MAT;
    float* agg  = (float*)(w + o); o += F32MAT;
    float* P    = (float*)(w + o); o += F32MAT;
    float* Q    = (float*)(w + o); o += F32MAT;
    float* y1u  = (float*)(w + o); o += F32MAT;
    float* y2u  = (float*)(w + o); o += F32MAT;
    unsigned short* hhi = (unsigned short*)(w + o); o += (size_t)NN * 64 * 2;
    unsigned short* hlo = (unsigned short*)(w + o); o += (size_t)NN * 64 * 2;
    float* sq   = (float*)(w + o); o += (size_t)NN * sizeof(float);
    int* knnidx = (int*)(w + o);   o += ((size_t)NN * KNN * sizeof(int) + 255) & ~(size_t)255;
    float* part = (float*)(w + o); o += (size_t)2048 * 128 * sizeof(float);
    float* bnpA = (float*)(w + o); o += 512;
    float* bnpB = (float*)(w + o); o += 512;
    float* poolp= (float*)(w + o); o += (size_t)256 * 64 * sizeof(float);

    k_lin_in<<<NN * 64 / 256, 256, 0, stream>>>(pos, linW, linB, h);

    for (int l = 0; l < 4; ++l) {
        const float* mW1 = msgW1 + l * 128 * 64;
        const float* mW2 = msgW2 + l * 64 * 64;
        const float* uW1 = updW1 + l * 128 * 64;
        const float* uW2 = updW2 + l * 64 * 64;

        k_prep<<<NN / 8, 256, 0, stream>>>(h, hhi, hlo, sq);
        k_knn<<<NN / 64, 64, 0, stream>>>(hhi, hlo, sq, knnidx);

        // ---- message MLP
        k_pq<<<NN / 16, 64, 0, stream>>>(h, mW1, P, Q);
        k_mstats1<<<512, 64, 0, stream>>>(P, Q, knnidx, msgB1 + l * 64, part);
        k_reduce_bn<<<1, 64, 0, stream>>>(part, 512, (float)EE,
                                          msgG1 + l * 64, msgBe1 + l * 64, bnpA);
        k_msg_pass<<<EE / 256, 256, 0, stream>>>(P, Q, knnidx, msgB1 + l * 64, bnpA,
                                                 mW2, msgB2 + l * 64, bnpB, part, agg, 0);
        k_reduce_bn<<<1, 64, 0, stream>>>(part, EE / 256, (float)EE,
                                          msgG2 + l * 64, msgBe2 + l * 64, bnpB);
        k_zero<<<NN * 64 / 256, 256, 0, stream>>>(agg, NN * 64);
        k_msg_pass<<<EE / 256, 256, 0, stream>>>(P, Q, knnidx, msgB1 + l * 64, bnpA,
                                                 mW2, msgB2 + l * 64, bnpB, part, agg, 1);

        // ---- update MLP + residual
        k_ustats1<<<512, 64, 0, stream>>>(h, agg, uW1, updB1 + l * 64, y1u, part);
        k_reduce_bn<<<1, 64, 0, stream>>>(part, 512, (float)NN,
                                          updG1 + l * 64, updBe1 + l * 64, bnpA);
        k_ustats2<<<NN / 64, 256, 0, stream>>>(y1u, bnpA, uW2, updB2 + l * 64, y2u, part);
        k_reduce_bn<<<1, 64, 0, stream>>>(part, NN / 64, (float)NN,
                                          updG2 + l * 64, updBe2 + l * 64, bnpB);
        k_ufinal<<<NN * 64 / 256, 256, 0, stream>>>(h, y2u, bnpB);
    }

    k_pool1<<<NN / 64, 64, 0, stream>>>(h, poolp);
    k_pool2<<<1, 64, 0, stream>>>(poolp, predW, predB, (float*)d_out);
}
